// MMSSL_29850022707359
// MI455X (gfx1250) — matrix operand order repaired and checked by reading, not compiled, not run on hardware
//
#include <hip/hip_runtime.h>
#include <cstdint>
#include <cstddef>

// ---------------------------------------------------------------------------
// MMSSL forward for MI455X (gfx1250, wave32).
// All big matmuls are C[M,64] = A[M,K] @ B[K,64] -> v_wmma_f32_16x16x32_bf16.
// ---------------------------------------------------------------------------

typedef __attribute__((ext_vector_type(16))) __bf16        bf16x16;
typedef __attribute__((ext_vector_type(8)))  float         f32x8;
typedef __attribute__((ext_vector_type(4)))  float         f32x4;
typedef __attribute__((ext_vector_type(4)))  unsigned int  u32x4;

union BFrag {
  bf16x16 v;
  u32x4 q[2];
  __bf16 h[16];
};

// f32 -> bf16 (RNE via hardware convert)
__device__ __forceinline__ unsigned short bf_bits(float f) {
  union { __bf16 h; unsigned short u; } c;
  c.h = (__bf16)f;
  return c.u;
}

// ----------------------------- GEMM ----------------------------------------
// C[M,64] = A[M,K] @ B[K,64] (+ optional bias[64]).
// Bt = B transposed to [64][K] in bf16. A is f32 (converted on the fly) or bf16.
// NT: stream A with non-temporal loads (single-use 128MB graphs).
// WB: while converting f32 A, also write the bf16 copy of A to Abf_out
//     (done by the t==0 col-tile waves only; each element stored exactly once).
// Wave computes one 16x16 tile across full K, double-buffered K loop.
// Block: 8 waves = 2 row strips x 4 col tiles -> 32 rows/block.
// Requires M % 32 == 0, K % 32 == 0, K >= 64.
template<bool A_BF16, bool NT, bool WB>
__global__ __launch_bounds__(256)
void gemm_n64(const void* __restrict__ A_, unsigned short* __restrict__ Abf_out,
              const unsigned short* __restrict__ Bt,
              const float* __restrict__ bias, float* __restrict__ C, int M, int K) {
  const int wave  = threadIdx.x >> 5;
  const int lane  = threadIdx.x & 31;
  const int r     = lane & 15;
  const int hi    = lane >> 4;
  const int strip = wave >> 2;     // 0..1
  const int t     = wave & 3;      // col tile 0..3
  const int row   = blockIdx.x * 32 + strip * 16 + r;

  const unsigned short* bp = Bt + (size_t)(t * 16 + r) * (size_t)K;

  auto loadA = [&](BFrag& a, int kc) {
    if constexpr (A_BF16) {
      const unsigned short* ap =
          (const unsigned short*)A_ + (size_t)row * (size_t)K + kc + hi * 8;
      a.q[0] = *(const u32x4*)(ap);        // K = kc+hi*8 .. +8
      a.q[1] = *(const u32x4*)(ap + 16);   // K = kc+16+hi*8 .. +8
    } else {
      const float* ap = (const float*)A_ + (size_t)row * (size_t)K + kc + hi * 8;
      f32x4 f0, f1, f2, f3;
      if constexpr (NT) {
        f0 = __builtin_nontemporal_load((const f32x4*)(ap));
        f1 = __builtin_nontemporal_load((const f32x4*)(ap + 4));
        f2 = __builtin_nontemporal_load((const f32x4*)(ap + 16));
        f3 = __builtin_nontemporal_load((const f32x4*)(ap + 20));
      } else {
        f0 = *(const f32x4*)(ap);
        f1 = *(const f32x4*)(ap + 4);
        f2 = *(const f32x4*)(ap + 16);
        f3 = *(const f32x4*)(ap + 20);
      }
#pragma unroll
      for (int j = 0; j < 4; ++j) a.h[j]      = (__bf16)f0[j];
#pragma unroll
      for (int j = 0; j < 4; ++j) a.h[4 + j]  = (__bf16)f1[j];
#pragma unroll
      for (int j = 0; j < 4; ++j) a.h[8 + j]  = (__bf16)f2[j];
#pragma unroll
      for (int j = 0; j < 4; ++j) a.h[12 + j] = (__bf16)f3[j];
      if constexpr (WB) {
        if (t == 0) {  // wave-uniform branch; EXEC stays all-ones
          unsigned short* op = Abf_out + (size_t)row * (size_t)K + kc + hi * 8;
          *(u32x4*)(op)      = a.q[0];
          *(u32x4*)(op + 16) = a.q[1];
        }
      }
    }
  };
  auto loadB = [&](BFrag& b, int kc) {
    const unsigned short* bkp = bp + kc + hi * 8;  // col n=t*16+r, K=kc+hi*8..+8, kc+16+hi*8..+8
    b.q[0] = *(const u32x4*)(bkp);
    b.q[1] = *(const u32x4*)(bkp + 16);
  };

  f32x8 acc = {};
  BFrag a0, b0;
  loadA(a0, 0);
  loadB(b0, 0);
  for (int kc = 32; kc < K; kc += 32) {
    BFrag a1, b1;
    loadA(a1, kc);                 // prefetch next chunk
    loadB(b1, kc);
    acc = __builtin_amdgcn_wmma_f32_16x16x32_bf16(
        false, a0.v, false, b0.v, (short)0, acc, false, false);
    a0 = a1;
    b0 = b1;
  }
  acc = __builtin_amdgcn_wmma_f32_16x16x32_bf16(
      false, a0.v, false, b0.v, (short)0, acc, false, false);

  const int mbase = blockIdx.x * 32 + strip * 16 + hi * 8;
  const int col   = t * 16 + r;
  float bv = bias ? bias[col] : 0.0f;
#pragma unroll
  for (int j = 0; j < 8; ++j) {
    C[(size_t)(mbase + j) * 64 + col] = acc[j] + bv;
  }
}

// --------------------- B transpose: f32 [K,64] -> bf16 [64,K] ---------------
__global__ __launch_bounds__(256)
void transpose_bf16_64(const float* __restrict__ B, unsigned short* __restrict__ Bt, int K) {
  int k  = blockIdx.x * 64 + (threadIdx.x & 63);
  int n0 = threadIdx.x >> 6;
  if (k >= K) return;
  for (int n = n0; n < 64; n += 4)
    Bt[(size_t)n * (size_t)K + k] = bf_bits(B[(size_t)k * 64 + n]);
}

// --------------------- wave helpers -----------------------------------------
__device__ __forceinline__ float wave_sum32(float x) {
#pragma unroll
  for (int off = 16; off >= 1; off >>= 1) x += __shfl_xor(x, off);
  return x;
}
__device__ __forceinline__ float wave_max32(float x) {
#pragma unroll
  for (int off = 16; off >= 1; off >>= 1) x = fmaxf(x, __shfl_xor(x, off));
  return x;
}

// --------------------- row softmax over 64 cols (in place) ------------------
__global__ __launch_bounds__(256)
void row_softmax64(float* __restrict__ X, int M) {
  int row  = blockIdx.x * 8 + (threadIdx.x >> 5);
  int lane = threadIdx.x & 31;
  if (row >= M) return;
  float* p = X + (size_t)row * 64;
  float a = p[lane], b = p[lane + 32];
  float m = wave_max32(fmaxf(a, b));
  float ea = __expf(a - m), eb = __expf(b - m);
  float s = wave_sum32(ea + eb);
  p[lane] = ea / s;
  p[lane + 32] = eb / s;
}

// --------------------- attention combine ------------------------------------
// Q,K: [2,N,64] (head h = cols h*16..h*16+16). Y: [2,4,N,64] = x_b @ Wc_h.
// g[n,:] = id_emb[n,:] + 0.36 * norm( 0.5 * sum_a sum_{h,b} s[h,a,b,n]*Y[b,h,n,:] )
__global__ __launch_bounds__(256)
void attn_combine(const float* __restrict__ Q, const float* __restrict__ Km,
                  const float* __restrict__ Y, const float* __restrict__ id_emb,
                  float* __restrict__ g, int N) {
  int n    = blockIdx.x * 8 + (threadIdx.x >> 5);
  int lane = threadIdx.x & 31;
  if (n >= N) return;
  int t = lane & 15;                 // (h,a,b) combo; lanes 16..31 mirror 0..15
  int h = t >> 2, a = (t >> 1) & 1, b = t & 1;
  const float* qp = Q  + ((size_t)a * N + n) * 64 + h * 16;
  const float* kp = Km + ((size_t)b * N + n) * 64 + h * 16;
  float dot = 0.f;
#pragma unroll
  for (int d = 0; d < 16; ++d) dot += qp[d] * kp[d];
  dot *= 0.25f;                       // 1/sqrt(dh), dh=16
  float other = __shfl_xor(dot, 1);   // partner b
  float mx = fmaxf(dot, other);
  float e  = __expf(dot - mx);
  float eo = __shfl_xor(e, 1);
  float s  = e / (e + eo);            // softmax over key modality

  float acc0 = 0.f, acc1 = 0.f;
#pragma unroll
  for (int tt = 0; tt < 16; ++tt) {
    float sv = __shfl(s, tt);
    int hh = tt >> 2, bb = tt & 1;
    const float* y = Y + ((size_t)(bb * 4 + hh) * N + n) * 64;
    acc0 += sv * y[lane];
    acc1 += sv * y[lane + 32];
  }
  acc0 *= 0.5f; acc1 *= 0.5f;        // mean over modality a
  float nrm = fmaxf(sqrtf(wave_sum32(acc0 * acc0 + acc1 * acc1)), 1e-12f);
  size_t base = (size_t)n * 64;
  g[base + lane]      = id_emb[base + lane]      + 0.36f * acc0 / nrm;
  g[base + lane + 32] = id_emb[base + lane + 32] + 0.36f * acc1 / nrm;
}

// --------------------- final combine ----------------------------------------
// out = mean(g0,g1,g2) + 0.55*rownorm(f1) + 0.55*rownorm(f2); duplicated store.
__global__ __launch_bounds__(256)
void final_combine(const float* __restrict__ g0, const float* __restrict__ g1,
                   const float* __restrict__ g2, const float* __restrict__ f1,
                   const float* __restrict__ f2, float* __restrict__ outA,
                   float* __restrict__ outB, int M) {
  int row  = blockIdx.x * 8 + (threadIdx.x >> 5);
  int lane = threadIdx.x & 31;
  if (row >= M) return;
  size_t base = (size_t)row * 64;
  float fa1 = f1[base + lane], fb1 = f1[base + lane + 32];
  float n1 = fmaxf(sqrtf(wave_sum32(fa1 * fa1 + fb1 * fb1)), 1e-12f);
  float fa2 = f2[base + lane], fb2 = f2[base + lane + 32];
  float n2 = fmaxf(sqrtf(wave_sum32(fa2 * fa2 + fb2 * fb2)), 1e-12f);
  const float inv3 = 1.0f / 3.0f;
  float va = (g0[base + lane] + g1[base + lane] + g2[base + lane]) * inv3
             + 0.55f * (fa1 / n1 + fa2 / n2);
  float vb = (g0[base + lane + 32] + g1[base + lane + 32] + g2[base + lane + 32]) * inv3
             + 0.55f * (fb1 / n1 + fb2 / n2);
  outA[base + lane] = va; outA[base + lane + 32] = vb;
  outB[base + lane] = va; outB[base + lane + 32] = vb;
}

// ---------------------------------------------------------------------------
extern "C" void kernel_launch(void* const* d_in, const int* in_sizes, int n_in,
                              void* d_out, int out_size, void* d_ws, size_t ws_size,
                              hipStream_t stream) {
  (void)in_sizes; (void)n_in; (void)out_size;
  const size_t NU = 8192, NI = 4096;

  const float* ui_graph    = (const float*)d_in[0];
  const float* iu_graph    = (const float*)d_in[1];
  const float* img_ui      = (const float*)d_in[2];
  const float* img_iu      = (const float*)d_in[3];
  const float* txt_ui      = (const float*)d_in[4];
  const float* txt_iu      = (const float*)d_in[5];
  const float* image_feats = (const float*)d_in[6];
  const float* text_feats  = (const float*)d_in[7];
  const float* W_img       = (const float*)d_in[8];
  const float* b_img       = (const float*)d_in[9];
  const float* W_txt       = (const float*)d_in[10];
  const float* b_txt       = (const float*)d_in[11];
  const float* user_id_emb = (const float*)d_in[12];
  const float* item_id_emb = (const float*)d_in[13];
  const float* w_q         = (const float*)d_in[14];
  const float* w_k         = (const float*)d_in[15];
  const float* w_cat       = (const float*)d_in[16];

  // ---- output slices (tuple concatenated flat, f32) ----
  float* out    = (float*)d_out;
  float* o_ug   = out;                    // u_g            [8192,64]
  float* o_ig   = o_ug   + NU * 64;       // i_g            [4096,64]
  float* o_iif  = o_ig   + NI * 64;       // image_item_feats
  float* o_tif  = o_iif  + NI * 64;       // text_item_feats
  float* o_iuf  = o_tif  + NI * 64;       // image_user_feats
  float* o_tuf  = o_iuf  + NU * 64;       // text_user_feats
  float* o_ug2  = o_tuf  + NU * 64;       // u_g (dup)
  float* o_ig2  = o_ug2  + NU * 64;       // i_g (dup)
  float* o_iuid = o_ig2  + NI * 64;       // image_user_id  [8192,64]
  float* o_tuid = o_iuid + NU * 64;       // text_user_id   [8192,64]

  // ---- workspace layout (deterministic) ----
  char* wsb = (char*)d_ws;
  size_t off = 0;
  auto alloc = [&](size_t bytes) -> void* {
    void* p = wsb + off;
    off = (off + bytes + 255) & ~(size_t)255;
    return p;
  };

  bool use_bf = ws_size >= ((size_t)200 << 20);  // room for bf16 graph copies
  unsigned short* ui_bf = nullptr;
  unsigned short* iu_bf = nullptr;
  if (use_bf) {
    ui_bf = (unsigned short*)alloc(NU * NI * 2);
    iu_bf = (unsigned short*)alloc(NI * NU * 2);
  }
  float* img_f = (float*)alloc(NI * 64 * 4);
  float* txt_f = (float*)alloc(NI * 64 * 4);
  unsigned short* Bt_Wimg = (unsigned short*)alloc((size_t)64 * 4096 * 2);
  unsigned short* Bt_Wtxt = (unsigned short*)alloc((size_t)64 * 1024 * 2);
  unsigned short* Bt_imgf = (unsigned short*)alloc(64 * NI * 2);
  unsigned short* Bt_txtf = (unsigned short*)alloc(64 * NI * 2);
  unsigned short* Bt_item = (unsigned short*)alloc(64 * NI * 2);
  unsigned short* Bt_user = (unsigned short*)alloc(64 * NU * 2);
  unsigned short* Bt_iuf  = (unsigned short*)alloc(64 * NU * 2);
  unsigned short* Bt_tuf  = (unsigned short*)alloc(64 * NU * 2);
  unsigned short* Bt_wq   = (unsigned short*)alloc(64 * 64 * 2);
  unsigned short* Bt_wk   = (unsigned short*)alloc(64 * 64 * 2);
  unsigned short* Bt_wc   = (unsigned short*)alloc(4 * 64 * 64 * 2);
  float* Xi  = (float*)alloc(2 * NI * 64 * 4);   // [image_item_id ; text_item_id]
  float* Qu  = (float*)alloc(2 * NU * 64 * 4);
  float* Ku_ = (float*)alloc(2 * NU * 64 * 4);
  float* Qi  = (float*)alloc(2 * NI * 64 * 4);
  float* Ki  = (float*)alloc(2 * NI * 64 * 4);
  float* Yu  = (float*)alloc(8 * NU * 64 * 4);
  float* Yi  = (float*)alloc(8 * NI * 64 * 4);
  float* u0 = (float*)alloc(NU * 64 * 4);
  float* u1 = (float*)alloc(NU * 64 * 4);
  float* u2 = (float*)alloc(NU * 64 * 4);
  float* i0 = (float*)alloc(NI * 64 * 4);
  float* i1 = (float*)alloc(NI * 64 * 4);
  float* i2 = (float*)alloc(NI * 64 * 4);
  unsigned short* Bt_i0 = (unsigned short*)alloc(64 * NI * 2);
  unsigned short* Bt_u1 = (unsigned short*)alloc(64 * NU * 2);
  unsigned short* Bt_i1 = (unsigned short*)alloc(64 * NI * 2);
  unsigned short* Bt_u2 = (unsigned short*)alloc(64 * NU * 2);

  // ---- launch helpers ----
  auto gemmF = [&](const float* A, const unsigned short* Bt, const float* bias,
                   float* C, int M, int K) {       // small / reused A: RT loads
    gemm_n64<false, false, false><<<dim3(M / 32), dim3(256), 0, stream>>>(
        (const void*)A, nullptr, Bt, bias, C, M, K);
  };
  auto gemmFnt = [&](const float* A, const unsigned short* Bt, const float* bias,
                     float* C, int M, int K) {     // huge single-use A: NT loads
    gemm_n64<false, true, false><<<dim3(M / 32), dim3(256), 0, stream>>>(
        (const void*)A, nullptr, Bt, bias, C, M, K);
  };
  auto gemmFwb = [&](const float* A, unsigned short* Abf, const unsigned short* Bt,
                     float* C, int M, int K) {     // first ui/iu use: NT + bf16 writeback
    gemm_n64<false, true, true><<<dim3(M / 32), dim3(256), 0, stream>>>(
        (const void*)A, Abf, Bt, nullptr, C, M, K);
  };
  auto gemmB = [&](const unsigned short* A, const unsigned short* Bt, float* C, int M, int K) {
    gemm_n64<true, false, false><<<dim3(M / 32), dim3(256), 0, stream>>>(
        (const void*)A, nullptr, Bt, nullptr, C, M, K);
  };
  auto gemmG = [&](const unsigned short* Abf, const float* Af, const unsigned short* Bt,
                   float* C, int M, int K) {  // later graph uses: bf16 copy if available
    if (use_bf) gemmB(Abf, Bt, C, M, K); else gemmF(Af, Bt, nullptr, C, M, K);
  };
  auto transp = [&](const float* B, unsigned short* Bt, int K) {
    transpose_bf16_64<<<dim3((K + 63) / 64), dim3(256), 0, stream>>>(B, Bt, K);
  };

  // ---- phase 0: static transposes ----
  transp(W_img, Bt_Wimg, 4096);
  transp(W_txt, Bt_Wtxt, 1024);
  transp(item_id_emb, Bt_item, (int)NI);
  transp(user_id_emb, Bt_user, (int)NU);
  transp(w_q, Bt_wq, 64);
  transp(w_k, Bt_wk, 64);
  for (int h = 0; h < 4; ++h) transp(w_cat + h * 64 * 64, Bt_wc + h * 64 * 64, 64);

  // ---- phase 1: feature projections ----
  gemmFnt(image_feats, Bt_Wimg, b_img, img_f, (int)NI, 4096);
  gemmFnt(text_feats,  Bt_Wtxt, b_txt, txt_f, (int)NI, 1024);
  transp(img_f, Bt_imgf, (int)NI);
  transp(txt_f, Bt_txtf, (int)NI);

  // ---- phase 2: graph propagation ----
  // First use of ui_graph/iu_graph fuses the f32->bf16 caching into the GEMM.
  if (use_bf) gemmFwb(ui_graph, ui_bf, Bt_imgf, o_iuf, (int)NU, (int)NI);
  else        gemmF  (ui_graph,        Bt_imgf, nullptr, o_iuf, (int)NU, (int)NI);
  gemmG(ui_bf, ui_graph, Bt_txtf, o_tuf, (int)NU, (int)NI);   // text_user_feats
  transp(o_iuf, Bt_iuf, (int)NU);
  transp(o_tuf, Bt_tuf, (int)NU);
  if (use_bf) gemmFwb(iu_graph, iu_bf, Bt_iuf, o_iif, (int)NI, (int)NU);
  else        gemmF  (iu_graph,        Bt_iuf, nullptr, o_iif, (int)NI, (int)NU);
  gemmG(iu_bf, iu_graph, Bt_tuf, o_tif, (int)NI, (int)NU);    // text_item_feats
  gemmFnt(img_ui, Bt_item, nullptr, o_iuid, (int)NU, (int)NI);  // image_user_id
  gemmFnt(txt_ui, Bt_item, nullptr, o_tuid, (int)NU, (int)NI);  // text_user_id
  gemmFnt(img_iu, Bt_user, nullptr, Xi,            (int)NI, (int)NU);  // image_item_id
  gemmFnt(txt_iu, Bt_user, nullptr, Xi + NI * 64,  (int)NI, (int)NU);  // text_item_id

  // ---- phase 3: attention (folded) ----
  const float* Xu = o_iuid;  // [2,NU,64] (image_user_id ; text_user_id contiguous)
  gemmF(Xu, Bt_wq, nullptr, Qu,  2 * (int)NU, 64);
  gemmF(Xu, Bt_wk, nullptr, Ku_, 2 * (int)NU, 64);
  gemmF(Xi, Bt_wq, nullptr, Qi,  2 * (int)NI, 64);
  gemmF(Xi, Bt_wk, nullptr, Ki,  2 * (int)NI, 64);
  for (int b = 0; b < 2; ++b)
    for (int h = 0; h < 4; ++h) {
      gemmF(Xu + (size_t)b * NU * 64, Bt_wc + h * 64 * 64, nullptr,
            Yu + (size_t)(b * 4 + h) * NU * 64, (int)NU, 64);
      gemmF(Xi + (size_t)b * NI * 64, Bt_wc + h * 64 * 64, nullptr,
            Yi + (size_t)(b * 4 + h) * NI * 64, (int)NI, 64);
    }
  attn_combine<<<dim3((unsigned)(NU / 8)), 256, 0, stream>>>(Qu, Ku_, Yu, user_id_emb, u0, (int)NU);
  attn_combine<<<dim3((unsigned)(NI / 8)), 256, 0, stream>>>(Qi, Ki,  Yi, item_id_emb, i0, (int)NI);

  // ---- phase 4: N_UI propagation loop ----
  transp(i0, Bt_i0, (int)NI);
  gemmG(ui_bf, ui_graph, Bt_i0, u1, (int)NU, (int)NI);
  transp(u1, Bt_u1, (int)NU);
  gemmG(iu_bf, iu_graph, Bt_u1, i1, (int)NI, (int)NU);
  transp(i1, Bt_i1, (int)NI);
  gemmG(ui_bf, ui_graph, Bt_i1, u2, (int)NU, (int)NI);
  row_softmax64<<<dim3((unsigned)(NU / 8)), 256, 0, stream>>>(u2, (int)NU);
  transp(u2, Bt_u2, (int)NU);
  gemmG(iu_bf, iu_graph, Bt_u2, i2, (int)NI, (int)NU);
  row_softmax64<<<dim3((unsigned)(NI / 8)), 256, 0, stream>>>(i2, (int)NI);

  // ---- phase 5: final combine ----
  final_combine<<<dim3((unsigned)(NU / 8)), 256, 0, stream>>>(u0, u1, u2, o_iuf, o_tuf,
                                                              o_ug, o_ug2, (int)NU);
  final_combine<<<dim3((unsigned)(NI / 8)), 256, 0, stream>>>(i0, i1, i2, o_iif, o_tif,
                                                              o_ig, o_ig2, (int)NI);
}
